// ConvLayer_609885356795
// MI455X (gfx1250) — matrix operand order repaired and checked by reading, not compiled, not run on hardware
//
#include <hip/hip_runtime.h>
#include <hip/hip_bf16.h>

// ---------------- problem constants ----------------
#define NATOM 50000
#define MNBR  12
#define AFEA  64        // A
#define BFEA  41        // B
#define NBRPAD 48       // edge features padded to 48 bf16 for aligned copies
#define KDIM  169       // 2A + B
#define KPAD  192       // padded K (6 x 32)
#define ODIM  128       // 2A output channels
#define BN1CNT 600000   // N*M rows for bn1
#define BN_EPS 1e-5f

typedef __attribute__((ext_vector_type(16))) __bf16 v16bf;
typedef __attribute__((ext_vector_type(8)))  float  v8f;
typedef __attribute__((ext_vector_type(4)))  unsigned int u32x4;
typedef __attribute__((ext_vector_type(8)))  int          i32x8;
typedef __attribute__((ext_vector_type(4)))  int          i32x4;

// ---------------- helpers ----------------
__device__ __forceinline__ unsigned short f2bf(float f) {
  // round-to-nearest-even f32 -> bf16
  unsigned int u = __float_as_uint(f);
  unsigned int r = u + 0x7FFFu + ((u >> 16) & 1u);
  return (unsigned short)(r >> 16);
}

struct Bf16x16Bits { uint4 lo; uint4 hi; };
static_assert(sizeof(Bf16x16Bits) == 32, "frag size");

// Load a 16-element bf16 fragment as two 128-bit LDS loads.
__device__ __forceinline__ v16bf ld_frag(const unsigned short* base, int off0, int off1) {
  Bf16x16Bits s;
  s.lo = *reinterpret_cast<const uint4*>(base + off0);
  s.hi = *reinterpret_cast<const uint4*>(base + off1);
  return __builtin_bit_cast(v16bf, s);
}

__device__ __forceinline__ float softplusf(float x) {
  return fmaxf(x, 0.f) + log1pf(__expf(-fabsf(x)));   // stable, branch-free
}
__device__ __forceinline__ float sigmoidf(float x) {
  return 1.f / (1.f + __expf(-x));
}

// ---- Tensor Data Mover: DMA the 128x192 bf16 weight tile from global into LDS.
// D# group0: count=1, lds_addr, 57-bit global addr, type=2.
// D# group1: data_size=2B, tensor_dim0=192, tensor_dim1=128, tile=192x128, stride0=192.
__device__ __forceinline__ void tdm_load_W(const unsigned short* Wb, unsigned short* ldsW) {
  unsigned long long ga = (unsigned long long)Wb;
  unsigned ldsa = (unsigned)(unsigned long long)(void*)ldsW;
  u32x4 g0;
  g0[0] = 1u;                                   // count = 1 valid descriptor
  g0[1] = ldsa;                                 // lds_addr [63:32]
  g0[2] = (unsigned)ga;                         // global_addr low
  g0[3] = (unsigned)(ga >> 32) | 0x80000000u;   // global_addr hi | type=2 ("image")
  i32x8 g1;
  g1[0] = 0x00010000;                           // data_size = 1 (2 bytes)
  g1[1] = (int)(192u << 16);                    // tensor_dim0 = 192 (low16 @ bits 63:48)
  g1[2] = (int)(128u << 16);                    // tensor_dim1 = 128 (low16 @ bits 95:80)
  g1[3] = (int)(192u << 16);                    // tile_dim0   = 192 (bits 127:112)
  g1[4] = 128;                                  // tile_dim1   = 128 (bits 143:128)
  g1[5] = 192;                                  // tensor_dim0_stride = 192
  g1[6] = 0; g1[7] = 0;
  i32x4 z4 = {};
#if defined(__clang_major__) && (__clang_major__ >= 23)
  i32x8 z8 = {};
  __builtin_amdgcn_tensor_load_to_lds(g0, g1, z4, z4, z8, 0);
#else
  __builtin_amdgcn_tensor_load_to_lds(g0, g1, z4, z4, 0);
#endif
}

// Stage 16 gathered+concatenated bf16 feature rows into LDS: pure aligned 16B copies.
// Row layout (ushorts): [0,64) self | [64,128) gathered neighbor | [128,176) padded edge | [176,192) zero
__device__ __forceinline__ void stage_rows_bf(const unsigned short* __restrict__ atomb,
                                              const unsigned short* __restrict__ nbrb,
                                              const int* __restrict__ idx,
                                              int nBase, int m,
                                              unsigned short* ldsA, int* ldsN, int tid) {
  __syncthreads();   // protect previous iteration's readers of ldsA/ldsN
  if (tid < 16) ldsN[tid] = idx[(nBase + tid) * MNBR + m];
  __syncthreads();
  // 16 rows x 24 chunks of 8 ushorts (16B) = 384 chunks; 3 per thread
  for (int j = tid; j < 16 * 24; j += 128) {
    int r = j / 24, c8 = j - r * 24;
    int n = nBase + r;
    uint4 v = {0u, 0u, 0u, 0u};
    if (c8 < 8)       v = *(const uint4*)(atomb + n * AFEA + c8 * 8);
    else if (c8 < 16) v = *(const uint4*)(atomb + ldsN[r] * AFEA + (c8 - 8) * 8);
    else if (c8 < 22) v = *(const uint4*)(nbrb + (n * MNBR + m) * NBRPAD + (c8 - 16) * 8);
    *(uint4*)(ldsA + r * KPAD + c8 * 8) = v;
  }
  __syncthreads();
}

// ---------------- tiny kernels ----------------
__global__ void k_init(float* ws) {
  int i = blockIdx.x * blockDim.x + threadIdx.x;
  if (i < 768) ws[i] = 0.f;
}

__global__ void k_convW(const float* __restrict__ W, unsigned short* __restrict__ Wb) {
  int i = blockIdx.x * blockDim.x + threadIdx.x;
  if (i < ODIM * KPAD) {
    int o = i / KPAD, k = i - o * KPAD;
    Wb[i] = (k < KDIM) ? f2bf(W[o * KDIM + k]) : (unsigned short)0;
  }
}

__global__ void k_conv_atom(const float* __restrict__ a, unsigned short* __restrict__ ab) {
  int i = blockIdx.x * blockDim.x + threadIdx.x;
  if (i < NATOM * AFEA) ab[i] = f2bf(a[i]);
}

__global__ void k_conv_nbr(const float* __restrict__ nb, unsigned short* __restrict__ nbb) {
  int i = blockIdx.x * blockDim.x + threadIdx.x;
  if (i < NATOM * MNBR * NBRPAD) {
    int r = i / NBRPAD, c = i - r * NBRPAD;
    nbb[i] = (c < BFEA) ? f2bf(nb[r * BFEA + c]) : (unsigned short)0;
  }
}

__global__ void k_fin_bn(const float* __restrict__ sum, const float* __restrict__ sq,
                         const float* __restrict__ g, const float* __restrict__ bb,
                         float* __restrict__ scale, float* __restrict__ shift,
                         int nch, float invCnt) {
  int c = blockIdx.x * blockDim.x + threadIdx.x;
  if (c < nch) {
    float mean = sum[c] * invCnt;
    float var  = sq[c] * invCnt - mean * mean;
    float rstd = rsqrtf(var + BN_EPS);
    float sc   = g[c] * rstd;
    scale[c] = sc;
    shift[c] = bb[c] - mean * sc;
  }
}

// ---------------- pass A: gather + bf16 WMMA GEMM + bn1 statistics ----------------
__global__ __launch_bounds__(128) void k_gemm_stats(
    const unsigned short* __restrict__ atomb, const unsigned short* __restrict__ nbrb,
    const int* __restrict__ idx, const unsigned short* __restrict__ Wb,
    const float* __restrict__ bias, float* __restrict__ bn1sum, float* __restrict__ bn1sq) {
  __shared__ unsigned short ldsW[ODIM * KPAD];   // 48 KB, filled by TDM
  __shared__ unsigned short ldsA[16 * KPAD];     // 6 KB staged strip
  __shared__ int   ldsN[16];
  __shared__ float ldsStat[2 * ODIM];

  int tid = threadIdx.x;
  int lane = tid & 31, wv = tid >> 5;
  if (wv == 0) tdm_load_W(Wb, ldsW);            // async DMA, overlaps with setup
  for (int j = tid; j < 2 * ODIM; j += 128) ldsStat[j] = 0.f;

  int half16 = lane >> 4;
  int mrow   = lane & 15;
  int col0   = wv * 16 + mrow;                  // filter channel 0..63
  int col1   = col0 + 64;                       // core channel 64..127
  float bias0 = bias[col0], bias1 = bias[col1];

  if (wv == 0) __builtin_amdgcn_s_wait_tensorcnt(0);
  // stage_rows_bf's leading barrier publishes ldsW to all waves

  int nBase = blockIdx.x * 16;                  // 3125 blocks x 16 atoms
  for (int m = 0; m < MNBR; ++m) {
    stage_rows_bf(atomb, nbrb, idx, nBase, m, ldsA, ldsN, tid);
    v8f acc0 = {}; v8f acc1 = {};
#pragma unroll
    for (int ks = 0; ks < 6; ++ks) {
      int ka = ks * 32 + half16 * 16;           // A: 16 consecutive K per lane
      v16bf af = ld_frag(ldsA, mrow * KPAD + ka, mrow * KPAD + ka + 8);
      int kb = ks * 32 + half16 * 16;           // B: 16 consecutive K per lane
      v16bf b0 = ld_frag(ldsW, col0 * KPAD + kb, col0 * KPAD + kb + 8);
      v16bf b1 = ld_frag(ldsW, col1 * KPAD + kb, col1 * KPAD + kb + 8);
      acc0 = __builtin_amdgcn_wmma_f32_16x16x32_bf16(false, af, false, b0, (short)0, acc0, false, false);
      acc1 = __builtin_amdgcn_wmma_f32_16x16x32_bf16(false, af, false, b1, (short)0, acc1, false, false);
    }
    float s0 = 0.f, q0 = 0.f, s1 = 0.f, q1 = 0.f;
#pragma unroll
    for (int r = 0; r < 8; ++r) {
      float y0 = acc0[r] + bias0; s0 += y0; q0 += y0 * y0;
      float y1 = acc1[r] + bias1; s1 += y1; q1 += y1 * y1;
    }
    atomicAdd(&ldsStat[col0], s0); atomicAdd(&ldsStat[ODIM + col0], q0);
    atomicAdd(&ldsStat[col1], s1); atomicAdd(&ldsStat[ODIM + col1], q1);
  }
  __syncthreads();
  if (tid < ODIM) {
    atomicAdd(&bn1sum[tid], ldsStat[tid]);
    atomicAdd(&bn1sq[tid],  ldsStat[ODIM + tid]);
  }
}

// ---------------- pass C: recompute GEMM, bn1, gate, register M-reduction, bn2 stats ----------------
__global__ __launch_bounds__(128) void k_gemm_gate(
    const unsigned short* __restrict__ atomb, const unsigned short* __restrict__ nbrb,
    const int* __restrict__ idx, const unsigned short* __restrict__ Wb,
    const float* __restrict__ bias, const float* __restrict__ sc1,
    const float* __restrict__ sh1, float* __restrict__ nbrsum,
    float* __restrict__ bn2sum, float* __restrict__ bn2sq) {
  __shared__ unsigned short ldsW[ODIM * KPAD];
  __shared__ unsigned short ldsA[16 * KPAD];
  __shared__ int   ldsN[16];
  __shared__ float ldsStat[2 * AFEA];

  int tid = threadIdx.x;
  int lane = tid & 31, wv = tid >> 5;
  if (wv == 0) tdm_load_W(Wb, ldsW);
  for (int j = tid; j < 2 * AFEA; j += 128) ldsStat[j] = 0.f;

  int half16 = lane >> 4;
  int mrow   = lane & 15;
  int col0   = wv * 16 + mrow;                  // filter channel == output feature a
  int col1   = col0 + 64;                       // paired core channel, SAME wave
  float bias0 = bias[col0], bias1 = bias[col1];
  float scf = sc1[col0], shf = sh1[col0];
  float scc = sc1[col1], shc = sh1[col1];

  if (wv == 0) __builtin_amdgcn_s_wait_tensorcnt(0);

  int nBase = blockIdx.x * 16;
  v8f msum = {};                                // sum over M kept in registers
  for (int m = 0; m < MNBR; ++m) {
    stage_rows_bf(atomb, nbrb, idx, nBase, m, ldsA, ldsN, tid);
    v8f acc0 = {}; v8f acc1 = {};
#pragma unroll
    for (int ks = 0; ks < 6; ++ks) {
      int ka = ks * 32 + half16 * 16;
      v16bf af = ld_frag(ldsA, mrow * KPAD + ka, mrow * KPAD + ka + 8);
      int kb = ks * 32 + half16 * 16;
      v16bf b0 = ld_frag(ldsW, col0 * KPAD + kb, col0 * KPAD + kb + 8);
      v16bf b1 = ld_frag(ldsW, col1 * KPAD + kb, col1 * KPAD + kb + 8);
      acc0 = __builtin_amdgcn_wmma_f32_16x16x32_bf16(false, af, false, b0, (short)0, acc0, false, false);
      acc1 = __builtin_amdgcn_wmma_f32_16x16x32_bf16(false, af, false, b1, (short)0, acc1, false, false);
    }
#pragma unroll
    for (int r = 0; r < 8; ++r) {
      float yf = (acc0[r] + bias0) * scf + shf;     // bn1 on filter
      float yc = (acc1[r] + bias1) * scc + shc;     // bn1 on core
      msum[r] += sigmoidf(yf) * softplusf(yc);
    }
  }
  float s = 0.f, q = 0.f;
#pragma unroll
  for (int r = 0; r < 8; ++r) {
    int n = nBase + half16 * 8 + r;                 // D row -> atom index
    nbrsum[n * AFEA + col0] = msum[r];
    s += msum[r]; q += msum[r] * msum[r];
  }
  atomicAdd(&ldsStat[col0], s);
  atomicAdd(&ldsStat[AFEA + col0], q);
  __syncthreads();
  if (tid < AFEA) {
    atomicAdd(&bn2sum[tid], ldsStat[tid]);
    atomicAdd(&bn2sq[tid],  ldsStat[AFEA + tid]);
  }
}

// ---------------- final: bn2 apply + residual softplus ----------------
__global__ void k_out(const float* __restrict__ atom, const float* __restrict__ nbrsum,
                      const float* __restrict__ sc2, const float* __restrict__ sh2,
                      float* __restrict__ out) {
  int i = blockIdx.x * blockDim.x + threadIdx.x;
  if (i < NATOM * AFEA) {
    int a = i & (AFEA - 1);
    float v = atom[i] + nbrsum[i] * sc2[a] + sh2[a];
    out[i] = softplusf(v);
  }
}

// ---------------- launcher ----------------
extern "C" void kernel_launch(void* const* d_in, const int* in_sizes, int n_in,
                              void* d_out, int out_size, void* d_ws, size_t ws_size,
                              hipStream_t stream) {
  const float* atom = (const float*)d_in[0];   // (N, 64)
  const float* nbr  = (const float*)d_in[1];   // (N, 12, 41)
  const int*   idx  = (const int*)  d_in[2];   // (N, 12)
  const float* W    = (const float*)d_in[3];   // (128, 169)
  const float* bias = (const float*)d_in[4];   // (128,)
  const float* g1   = (const float*)d_in[5];
  const float* b1   = (const float*)d_in[6];
  const float* g2   = (const float*)d_in[7];
  const float* b2   = (const float*)d_in[8];
  float* out = (float*)d_out;

  float* ws = (float*)d_ws;
  float* bn1sum = ws +   0;  float* bn1sq = ws + 128;
  float* sc1    = ws + 256;  float* sh1   = ws + 384;
  float* bn2sum = ws + 512;  float* bn2sq = ws + 576;
  float* sc2    = ws + 640;  float* sh2   = ws + 704;
  unsigned short* Wb = (unsigned short*)(ws + 768);              // 128*192 bf16
  float* nbrsum = ws + 13056;                                    // (N,64) f32
  unsigned short* atomb = (unsigned short*)(ws + 3213056);       // (N,64) bf16   (6.4 MB)
  unsigned short* nbrb  = (unsigned short*)(ws + 4813056);       // (N,12,48) bf16 (57.6 MB, L2-resident)

  k_init <<<3, 256, 0, stream>>>(ws);
  k_convW<<<(ODIM * KPAD + 255) / 256, 256, 0, stream>>>(W, Wb);
  k_conv_atom<<<(NATOM * AFEA + 255) / 256, 256, 0, stream>>>(atom, atomb);
  k_conv_nbr<<<(NATOM * MNBR * NBRPAD + 255) / 256, 256, 0, stream>>>(nbr, nbrb);

  k_gemm_stats<<<NATOM / 16, 128, 0, stream>>>(atomb, nbrb, idx, Wb, bias, bn1sum, bn1sq);
  k_fin_bn<<<1, 128, 0, stream>>>(bn1sum, bn1sq, g1, b1, sc1, sh1, ODIM, 1.f / (float)BN1CNT);
  k_gemm_gate<<<NATOM / 16, 128, 0, stream>>>(atomb, nbrb, idx, Wb, bias, sc1, sh1,
                                              nbrsum, bn2sum, bn2sq);
  k_fin_bn<<<1, 64, 0, stream>>>(bn2sum, bn2sq, g2, b2, sc2, sh2, AFEA, 1.f / (float)NATOM);
  k_out<<<(NATOM * AFEA + 255) / 256, 256, 0, stream>>>(atom, nbrsum, sc2, sh2, out);
}
